// LinformerSelfAttention_53936199303433
// MI455X (gfx1250) — matrix operand order repaired and checked by reading, not compiled, not run on hardware
//
#include <hip/hip_runtime.h>

// ---------------------------------------------------------------------------
// Linformer self-attention forward, bf16 WMMA pipeline for gfx1250 (MI455X).
// B=4, N=4096, DIM=1024, H=16, K=256, HD=64.  All GEMMs are NT (both operands
// row-major over the contraction dim) and run on v_wmma_f32_16x16x32_bf16.
// GEMM tiles are staged with GLOBAL_LOAD_ASYNC_TO_LDS_B128 (ASYNCcnt) through
// a 3-deep LDS ring: one barrier per K-step, two tiles always in flight.
// ---------------------------------------------------------------------------

typedef __attribute__((ext_vector_type(16))) __bf16 v16bf;
typedef __attribute__((ext_vector_type(8)))  __bf16 v8bf;
typedef __attribute__((ext_vector_type(4)))  __bf16 v4bf;
typedef __attribute__((ext_vector_type(8)))  float  v8f;
typedef __attribute__((ext_vector_type(4)))  float  v4f;
typedef int v4i_vs __attribute__((vector_size(16)));   // matches builtin param

constexpr int   C_B   = 4;
constexpr int   C_N   = 4096;
constexpr int   C_DIM = 1024;
constexpr int   C_H   = 16;
constexpr int   C_K   = 256;
constexpr int   C_HD  = 64;
constexpr float ATTN_SCALE = 0.125f;   // 1/sqrt(64)

#if __has_builtin(__builtin_amdgcn_global_load_async_to_lds_b128) && \
    __has_builtin(__builtin_amdgcn_s_wait_asynccnt)
#define HAS_ASYNC_LDS 1
#else
#define HAS_ASYNC_LDS 0
#endif

__device__ inline void ld_async16(const __bf16* g, __bf16* l) {
#if HAS_ASYNC_LDS
  __builtin_amdgcn_global_load_async_to_lds_b128(
      (__attribute__((address_space(1))) v4i_vs*)g,
      (__attribute__((address_space(3))) v4i_vs*)l, 0, 0);
#else
  *(v8bf*)l = *(const v8bf*)g;
#endif
}

#if HAS_ASYNC_LDS
#define WAIT_ASYNC(nconst) __builtin_amdgcn_s_wait_asynccnt(nconst)
#else
#define WAIT_ASYNC(nconst) ((void)0)
#endif

__device__ inline v8f wmma_bf16(v16bf a, v16bf b, v8f c) {
  return __builtin_amdgcn_wmma_f32_16x16x32_bf16(
      /*neg_a=*/false, a, /*neg_b=*/false, b,
      /*c_mod=*/(short)0, c, /*reuse_a=*/false, /*reuse_b=*/false);
}

__device__ inline v8f zero8() {
  v8f z;
#pragma unroll
  for (int i = 0; i < 8; ++i) z[i] = 0.0f;
  return z;
}

// ---------------------------------------------------------------------------
// fp32 -> bf16 conversion (vectorized x4); n is a multiple of 4.
// ---------------------------------------------------------------------------
__global__ __launch_bounds__(256) void cvt_f32_bf16(
    const float* __restrict__ s, __bf16* __restrict__ d, int n) {
  int i = (blockIdx.x * 256 + threadIdx.x) * 4;
  if (i < n) {
    v4f v = *(const v4f*)(s + i);
    v4bf o;
#pragma unroll
    for (int j = 0; j < 4; ++j) o[j] = (__bf16)v[j];
    *(v4bf*)(d + i) = o;
  }
}

// ---------------------------------------------------------------------------
// Generic NT bf16 GEMM: C[M,N] = A[M,Kd] * B[N,Kd]^T, fp32 accumulate.
// Block tile BM x BN, K-step 32, 8 waves, wave tile WM x WN.
// EPI: 0 = QKV scatter (q scaled, k/v transposed per head)
//      1 = bf16 row-major   out0[z*oStride + m*ldo + n]
//      2 = bf16 transposed  out0[z*oStride + n*ldo + m]
//      3 = fp32 + bias      outF[m*N + n] = acc + bias[n]
// Batch: A offset = (z % aMod)*aStride, B offset = z*bStride.
// ---------------------------------------------------------------------------
template <int BM, int BN, int WM, int WN, int EPI>
__global__ __launch_bounds__(256) void gemm_nt(
    const __bf16* __restrict__ Ag, const __bf16* __restrict__ Bg,
    int M, int Nc, int Kd, int aMod, long aStride, long bStride,
    __bf16* __restrict__ o0, __bf16* __restrict__ o1, __bf16* __restrict__ o2,
    long oStride, int ldo, float* __restrict__ outF,
    const float* __restrict__ bias) {
  static_assert((BM / WM) * (BN / WN) == 8, "8 waves per block");
  constexpr int LDK  = 40;                       // padded rows (80B)
  constexpr int NBUF = HAS_ASYNC_LDS ? 3 : 1;    // LDS ring depth
  constexpr int ACH  = (BM * 32 / 8) / 256;      // per-thread A chunks/tile
  constexpr int BCH  = (BN * 32 / 8) / 256;      // per-thread B chunks/tile
  constexpr int CH   = ACH + BCH;                // async ops per wave per tile
  __shared__ __align__(16) __bf16 As[NBUF][BM * LDK];
  __shared__ __align__(16) __bf16 Bs[NBUF][BN * LDK];

  const int tid = threadIdx.x;
  const int wave = tid >> 5;
  const int lane = tid & 31;
  const int l = lane & 15;
  const int hf = lane >> 4;
  constexpr int WAVES_M = BM / WM;
  const int wm = wave % WAVES_M;
  const int wn = wave / WAVES_M;
  const int m0 = blockIdx.x * BM;
  const int n0 = blockIdx.y * BN;
  const int z = blockIdx.z;

  const __bf16* A = Ag + (size_t)(z % aMod) * (size_t)aStride;
  const __bf16* B = Bg + (size_t)z * (size_t)bStride;

  constexpr int MI = WM / 16;
  constexpr int NI = WN / 16;
  v8f acc[MI][NI];
#pragma unroll
  for (int mi = 0; mi < MI; ++mi)
#pragma unroll
    for (int nj = 0; nj < NI; ++nj) acc[mi][nj] = zero8();

  auto stage = [&](int kt, int buf) {
#pragma unroll
    for (int i = 0; i < ACH; ++i) {
      int idx = tid + i * 256;
      int r = idx >> 2, c = idx & 3;
      ld_async16(&A[(size_t)(m0 + r) * Kd + kt + c * 8],
                 &As[buf][r * LDK + c * 8]);
    }
#pragma unroll
    for (int i = 0; i < BCH; ++i) {
      int idx = tid + i * 256;
      int r = idx >> 2, c = idx & 3;
      ld_async16(&B[(size_t)(n0 + r) * Kd + kt + c * 8],
                 &Bs[buf][r * LDK + c * 8]);
    }
  };

  auto compute = [&](int p) {
    // A fragments: 16x32 layout (lanes 0-15 rows, K 0..15/16..31 contiguous)
    v16bf af[MI];
#pragma unroll
    for (int mi = 0; mi < MI; ++mi) {
      const __bf16* pa = &As[p][(wm * WM + mi * 16 + l) * LDK];
      v8bf lo = *(const v8bf*)(pa + hf * 16);
      v8bf hj = *(const v8bf*)(pa + hf * 16 + 8);
#pragma unroll
      for (int j = 0; j < 8; ++j) { af[mi][j] = lo[j]; af[mi][8 + j] = hj[j]; }
    }
    // B fragments: 32x16 layout (lane = column, K 0..15/16..31 contiguous)
    v16bf bfv[NI];
#pragma unroll
    for (int nj = 0; nj < NI; ++nj)
      bfv[nj] = *(const v16bf*)&Bs[p][(wn * WN + nj * 16 + l) * LDK + hf * 16];
#pragma unroll
    for (int mi = 0; mi < MI; ++mi)
#pragma unroll
      for (int nj = 0; nj < NI; ++nj)
        acc[mi][nj] = wmma_bf16(af[mi], bfv[nj], acc[mi][nj]);
  };

#if HAS_ASYNC_LDS
  const int T = Kd / 32;
  stage(0, 0);
  if (T > 1) stage(32, 1);
  for (int t = 0; t < T - 1; ++t) {
    WAIT_ASYNC(CH);        // tile t resident (async loads complete in order)
    __syncthreads();       // ...for every wave; also fences iter t-1 readers
    if (t + 2 < T) stage((t + 2) * 32, (t + 2) % 3);  // overwrites buf (t-1)%3
    compute(t % 3);
  }
  WAIT_ASYNC(0);
  __syncthreads();
  compute((T - 1) % 3);
#else
  for (int kt = 0; kt < Kd; kt += 32) {
    stage(kt, 0);
    __syncthreads();
    compute(0);
    __syncthreads();
  }
#endif

  // Epilogue: C layout = lane(0-15)=col, VGPR r = row hf*8+r.
#pragma unroll
  for (int mi = 0; mi < MI; ++mi)
#pragma unroll
    for (int nj = 0; nj < NI; ++nj)
#pragma unroll
      for (int r = 0; r < 8; ++r) {
        int m = m0 + wm * WM + mi * 16 + hf * 8 + r;
        int n = n0 + wn * WN + nj * 16 + l;
        float v = acc[mi][nj][r];
        if constexpr (EPI == 0) {
          int b = m >> 12;          // m / 4096
          int nn = m & 4095;
          int sec = n >> 10;        // 0=q 1=k 2=v
          int e = n & 1023;
          int h = e >> 6, d = e & 63;
          size_t bh = (size_t)(b * C_H + h);
          if (sec == 0)
            o0[(bh * C_N + nn) * C_HD + d] = (__bf16)(v * ATTN_SCALE);
          else if (sec == 1)
            o1[(bh * C_HD + d) * C_N + nn] = (__bf16)v;   // k^T (B,H,HD,N)
          else
            o2[(bh * C_HD + d) * C_N + nn] = (__bf16)v;   // v^T (B,H,HD,N)
        } else if constexpr (EPI == 1) {
          o0[(size_t)z * oStride + (size_t)m * ldo + n] = (__bf16)v;
        } else if constexpr (EPI == 2) {
          o0[(size_t)z * oStride + (size_t)n * ldo + m] = (__bf16)v;
        } else {
          outF[(size_t)m * Nc + n] = v + bias[n];
        }
      }
}

// ---------------------------------------------------------------------------
// Fused attention: scores = (q*scale) @ kproj^T, softmax over K=256, out = P @ vproj.
// Block = one (b,h) x 128 query rows; 8 waves x 16 rows each.
// kproj (K x HD) async-staged into LDS (overlapped with q fragment loads);
// the same LDS is reused for vproj^T (HD x K), async-staged under the softmax.
// All 16 score C-tiles live in registers; P bounces through a per-wave LDS
// scratch to convert C-layout -> A-fragment layout for the PV WMMAs.
// ---------------------------------------------------------------------------
__global__ __launch_bounds__(256) void linf_attn(
    const __bf16* __restrict__ q, const __bf16* __restrict__ kproj,
    const __bf16* __restrict__ vprojT, __bf16* __restrict__ obuf) {
  constexpr int LDKV = 72;   // kproj rows: 64 + 8 pad (144B)
  constexpr int LDVT = 264;  // vproj^T rows: 256 + 8 pad (528B)
  constexpr int LDP = 72;    // P scratch rows: 64 + 8 pad
  __shared__ __align__(16) __bf16 kv[C_K * LDKV];      // >= 64*LDVT too
  __shared__ __align__(16) __bf16 pbuf[8][16 * LDP];

  const int tid = threadIdx.x;
  const int wave = tid >> 5;
  const int lane = tid & 31;
  const int l = lane & 15;
  const int hf = lane >> 4;
  const int z = blockIdx.y;             // b*H + h
  const int b = z >> 4, h = z & 15;
  const int n0 = blockIdx.x * 128 + wave * 16;

  // Stage kproj (256 x 64) -> kv (padded rows), async; overlap q loads below.
  const __bf16* kp = kproj + (size_t)z * (C_K * C_HD);
#pragma unroll
  for (int i = 0; i < 8; ++i) {
    int idx = tid + i * 256;            // 2048 chunks of 8 bf16
    int r = idx >> 3, c = idx & 7;
    ld_async16(&kp[r * C_HD + c * 8], &kv[r * LDKV + c * 8]);
  }

  // q A-fragments straight from global (q is pre-scaled by 1/sqrt(HD)).
  const __bf16* qr = q + ((size_t)z * C_N + (n0 + l)) * C_HD;
  v16bf qf[2];
#pragma unroll
  for (int ds = 0; ds < 2; ++ds) {
    v8bf lo = *(const v8bf*)(qr + ds * 32 + hf * 16);
    v8bf hj = *(const v8bf*)(qr + ds * 32 + hf * 16 + 8);
#pragma unroll
    for (int j = 0; j < 8; ++j) { qf[ds][j] = lo[j]; qf[ds][8 + j] = hj[j]; }
  }
  WAIT_ASYNC(0);
  __syncthreads();

  // Scores: 16 tiles of 16 columns, contraction over HD=64 (2 WMMAs each).
  v8f S[16];
#pragma unroll
  for (int t = 0; t < 16; ++t) {
    v8f a = zero8();
#pragma unroll
    for (int ds = 0; ds < 2; ++ds) {
      v16bf bfr = *(const v16bf*)&kv[(t * 16 + l) * LDKV + ds * 32 + hf * 16];
      a = wmma_bf16(qf[ds], bfr, a);
    }
    S[t] = a;
  }
  __syncthreads();  // everyone done reading kv

  // Reuse kv as vproj^T (64 x 256) buffer; async loads run under the softmax.
  const __bf16* vp = vprojT + (size_t)z * (C_HD * C_K);
#pragma unroll
  for (int i = 0; i < 8; ++i) {
    int idx = tid + i * 256;            // 2048 chunks of 8 bf16
    int r = idx >> 5, c = idx & 31;
    ld_async16(&vp[r * C_K + c * 8], &kv[r * LDVT + c * 8]);
  }

  // Softmax: row m = hf*8+r lives in VGPR r across 16 lanes of this half-wave.
  float inv[8];
#pragma unroll
  for (int r = 0; r < 8; ++r) {
    float m = -3.0e38f;
#pragma unroll
    for (int t = 0; t < 16; ++t) m = fmaxf(m, S[t][r]);
#pragma unroll
    for (int off = 1; off < 16; off <<= 1) m = fmaxf(m, __shfl_xor(m, off, 32));
    float s = 0.0f;
#pragma unroll
    for (int t = 0; t < 16; ++t) {
      float e = __expf(S[t][r] - m);
      S[t][r] = e;
      s += e;
    }
#pragma unroll
    for (int off = 1; off < 16; off <<= 1) s += __shfl_xor(s, off, 32);
    inv[r] = 1.0f / s;
  }
  WAIT_ASYNC(0);
  __syncthreads();  // vproj^T staged

  // out = P @ V: iterate 4 chunks of 64 keys; C-layout P -> LDS -> A-fragments.
  __bf16* pw = pbuf[wave];
  v8f o[4];
#pragma unroll
  for (int dj = 0; dj < 4; ++dj) o[dj] = zero8();

#pragma unroll
  for (int kc = 0; kc < 4; ++kc) {
#pragma unroll
    for (int t = 0; t < 4; ++t)
#pragma unroll
      for (int r = 0; r < 8; ++r)
        pw[(hf * 8 + r) * LDP + t * 16 + l] =
            (__bf16)(S[kc * 4 + t][r] * inv[r]);
    // (same-wave DS write->read; hardware keeps LDS ops in order)
    v16bf pf[2];
#pragma unroll
    for (int ks = 0; ks < 2; ++ks) {
      const __bf16* pr = pw + l * LDP + ks * 32;
      v8bf lo = *(const v8bf*)(pr + hf * 16);
      v8bf hj = *(const v8bf*)(pr + hf * 16 + 8);
#pragma unroll
      for (int j = 0; j < 8; ++j) { pf[ks][j] = lo[j]; pf[ks][8 + j] = hj[j]; }
    }
#pragma unroll
    for (int dj = 0; dj < 4; ++dj)
#pragma unroll
      for (int ks = 0; ks < 2; ++ks) {
        v16bf bfr = *(const v16bf*)&kv[(dj * 16 + l) * LDVT + kc * 64 +
                                       ks * 32 + hf * 16];
        o[dj] = wmma_bf16(pf[ks], bfr, o[dj]);
      }
  }

  // Write attention output as (B, N, DIM) with DIM = H*HD (ready for out-proj).
#pragma unroll
  for (int dj = 0; dj < 4; ++dj)
#pragma unroll
    for (int r = 0; r < 8; ++r) {
      int nn = n0 + hf * 8 + r;
      obuf[((size_t)b * C_N + nn) * C_DIM + h * C_HD + dj * 16 + l] =
          (__bf16)o[dj][r];
    }
}

// ---------------------------------------------------------------------------
// Host-side orchestration.
// ---------------------------------------------------------------------------
extern "C" void kernel_launch(void* const* d_in, const int* in_sizes, int n_in,
                              void* d_out, int out_size, void* d_ws,
                              size_t ws_size, hipStream_t stream) {
  const float* x    = (const float*)d_in[0];
  const float* Wqkv = (const float*)d_in[1];
  const float* E    = (const float*)d_in[2];
  const float* F    = (const float*)d_in[3];
  const float* Wout = (const float*)d_in[4];
  const float* bout = (const float*)d_in[5];

  char* ws = (char*)d_ws;
  size_t off = 0;
  auto alloc = [&](size_t bytes) -> char* {
    char* p = ws + off;
    off = (off + bytes + 255) & ~(size_t)255;
    return p;
  };
  const size_t NX = (size_t)C_B * C_N * C_DIM;           // 16.78M elems
  __bf16* xb    = (__bf16*)alloc(NX * 2);
  __bf16* wqkvb = (__bf16*)alloc((size_t)3 * C_DIM * C_DIM * 2);
  __bf16* eb    = (__bf16*)alloc((size_t)C_H * C_K * C_N * 2);
  __bf16* fb    = (__bf16*)alloc((size_t)C_H * C_K * C_N * 2);
  __bf16* woutb = (__bf16*)alloc((size_t)C_DIM * C_DIM * 2);
  __bf16* qb    = (__bf16*)alloc(NX * 2);
  __bf16* kTb   = (__bf16*)alloc(NX * 2);
  __bf16* vTb   = (__bf16*)alloc(NX * 2);
  __bf16* kpb   = (__bf16*)alloc((size_t)C_B * C_H * C_K * C_HD * 2);
  __bf16* vpTb  = (__bf16*)alloc((size_t)C_B * C_H * C_HD * C_K * 2);
  __bf16* ob    = xb;  // x's bf16 copy is dead after stage 1; reuse for attn out

  auto cvt = [&](const float* s, __bf16* d, size_t n) {
    cvt_f32_bf16<<<dim3((unsigned)((n / 4 + 255) / 256)), dim3(256), 0,
                   stream>>>(s, d, (int)n);
  };
  cvt(x, xb, NX);
  cvt(Wqkv, wqkvb, (size_t)3 * C_DIM * C_DIM);
  cvt(E, eb, (size_t)C_H * C_K * C_N);
  cvt(F, fb, (size_t)C_H * C_K * C_N);
  cvt(Wout, woutb, (size_t)C_DIM * C_DIM);

  // Stage 1: qkv = x @ Wqkv^T (16384 x 3072 x 1024); scatter q/kT/vT.
  gemm_nt<128, 128, 32, 64, 0><<<dim3(128, 24, 1), 256, 0, stream>>>(
      xb, wqkvb, C_B * C_N, 3 * C_DIM, C_DIM, 1, 0, 0,
      qb, kTb, vTb, 0, 0, nullptr, nullptr);

  // Stage 2a: kproj[b,h] = E_h (256x4096) @ kT[b,h] (64x4096)^T -> (K,HD).
  gemm_nt<128, 64, 32, 32, 1><<<dim3(2, 1, C_B * C_H), 256, 0, stream>>>(
      eb, kTb, C_K, C_HD, C_N, C_H, (long)C_K * C_N, (long)C_HD * C_N,
      kpb, nullptr, nullptr, (long)C_K * C_HD, C_HD, nullptr, nullptr);

  // Stage 2b: vproj, stored transposed (HD,K) for contiguous PV B-fragments.
  gemm_nt<128, 64, 32, 32, 2><<<dim3(2, 1, C_B * C_H), 256, 0, stream>>>(
      fb, vTb, C_K, C_HD, C_N, C_H, (long)C_K * C_N, (long)C_HD * C_N,
      vpTb, nullptr, nullptr, (long)C_HD * C_K, C_K, nullptr, nullptr);

  // Stage 3: fused scores + softmax + PV.
  linf_attn<<<dim3(C_N / 128, C_B * C_H), 256, 0, stream>>>(qb, kpb, vpTb, ob);

  // Stage 4: out = attn_out @ Wout^T + bout (fp32 result).
  gemm_nt<128, 128, 32, 64, 3><<<dim3(128, 8, 1), 256, 0, stream>>>(
      ob, woutb, C_B * C_N, C_DIM, C_DIM, 1, 0, 0,
      nullptr, nullptr, nullptr, 0, 0, (float*)d_out, bout);

  (void)in_sizes; (void)n_in; (void)out_size; (void)ws_size;
}
